// Decoder_35201551958776
// MI455X (gfx1250) — matrix operand order repaired and checked by reading, not compiled, not run on hardware
//
#include <hip/hip_runtime.h>

#define DEVI __device__ __forceinline__

typedef __attribute__((ext_vector_type(16))) __bf16 bf16x16;
typedef __attribute__((ext_vector_type(8)))  float  f32x8;

union FragBF {
    bf16x16 v;
    uint4   q[2];
};

DEVI unsigned short f2bf(float f) {
    unsigned int u = __float_as_uint(f);
    u += 0x7FFFu + ((u >> 16) & 1u);   // round-to-nearest-even
    return (unsigned short)(u >> 16);
}

// A-fragment (16x32 bf16, M x K): lane holds row = lane&15; VGPR0-3 hold
// K = kgroup*8 .. +7 (contiguous 16B), VGPR4-7 hold K = 16+kgroup*8 .. +7.
DEVI void load_a(FragBF& f, const unsigned short* base, int ldk, int row, int kk, int lane) {
    const unsigned short* p = base + (size_t)(row + (lane & 15)) * ldk + kk + ((lane >> 4) << 3);
    f.q[0] = *(const uint4*)p;
    f.q[1] = *(const uint4*)(p + 16);
}

// B-fragment (32x16 bf16, K x N) sourced from an [N, K] row-major buffer:
// lane holds col = lane&15; K = kk + (lane>>4)*8 .. +7 and kk + 16 + (lane>>4)*8 .. +7.
DEVI void load_b(FragBF& f, const unsigned short* base, int ldk, int col, int kk, int lane) {
    const unsigned short* p = base + (size_t)(col + (lane & 15)) * ldk + kk + ((lane >> 4) << 3);
    f.q[0] = *(const uint4*)p;
    f.q[1] = *(const uint4*)(p + 16);
}

DEVI f32x8 wmma_bf16(const FragBF& a, const FragBF& b, f32x8 c) {
    return __builtin_amdgcn_wmma_f32_16x16x32_bf16(false, a.v, false, b.v,
                                                   (short)0, c, false, false);
}

// ---------------------------------------------------------------------------
// Weight convert + transpose: src [K,N] f32 -> dst [N,K] bf16
// ---------------------------------------------------------------------------
__global__ void k_convT(const float* __restrict__ src, unsigned short* __restrict__ dst,
                        int K, int N) {
    int total = K * N;
    for (int idx = blockIdx.x * blockDim.x + threadIdx.x; idx < total;
         idx += gridDim.x * blockDim.x) {
        int k = idx / N, n = idx - k * N;
        dst[(size_t)n * K + k] = f2bf(src[idx]);
    }
}

__global__ void k_conv(const float* __restrict__ src, unsigned short* __restrict__ dst, int n) {
    for (int i = blockIdx.x * blockDim.x + threadIdx.x; i < n; i += gridDim.x * blockDim.x)
        dst[i] = f2bf(src[i]);
}

// ---------------------------------------------------------------------------
// Embedding + sinusoid positional encoding (positions 5..S+4)
// ---------------------------------------------------------------------------
__global__ void k_embed(const int* __restrict__ ids, const float* __restrict__ emb,
                        float* __restrict__ xF, unsigned short* __restrict__ xB) {
    int row = blockIdx.x;           // token index in [0, 4096)
    int s   = row & 1023;
    int t   = threadIdx.x;
    int id  = ids[row];
    float pos = (float)(s + 5);
#pragma unroll
    for (int u = 0; u < 2; u++) {
        int c = t + u * 256;
        float e   = (float)(2 * (c >> 1)) * (1.0f / 512.0f);
        float ang = pos * __powf(10000.0f, -e);
        float pe  = (c & 1) ? __cosf(ang) : __sinf(ang);
        float v   = emb[(size_t)id * 512 + c] + pe;
        xF[(size_t)row * 512 + c] = v;
        xB[(size_t)row * 512 + c] = f2bf(v);
    }
}

// ---------------------------------------------------------------------------
// bf16 WMMA GEMM: out[M,N] = A[M,K] @ Bt[N,K]^T + bias
// 256 threads = 8 waves as 4(M) x 2(N); wave tile 32x64 = 2x4 WMMA tiles
// (8 v_wmma per 6 fragment loads -> ~22 FLOP/byte of fragment traffic).
// outBmode: 0 none, 1 row-major bf16, 2 attention-V transposed [b,h,64,1024]
// ---------------------------------------------------------------------------
__global__ void k_gemm(const unsigned short* __restrict__ A,
                       const unsigned short* __restrict__ Bt,
                       const float* __restrict__ bias,
                       float* __restrict__ outF,
                       unsigned short* __restrict__ outB,
                       int M, int N, int K, int relu, int outBmode) {
    const int lane = threadIdx.x & 31;
    const int wave = threadIdx.x >> 5;
    const int wm = wave >> 1, wn = wave & 1;
    const int rowBase = blockIdx.x * 128 + wm * 32;
    const int colBase = blockIdx.y * 128 + wn * 64;
    (void)M;

    f32x8 acc[2][4];
#pragma unroll
    for (int i = 0; i < 2; i++)
#pragma unroll
        for (int j = 0; j < 4; j++) acc[i][j] = (f32x8){0.f, 0.f, 0.f, 0.f, 0.f, 0.f, 0.f, 0.f};

    for (int kk = 0; kk < K; kk += 32) {
        FragBF a0, a1, b0, b1, b2, b3;
        load_a(a0, A, K, rowBase, kk, lane);
        load_a(a1, A, K, rowBase + 16, kk, lane);
        load_b(b0, Bt, K, colBase, kk, lane);
        load_b(b1, Bt, K, colBase + 16, kk, lane);
        load_b(b2, Bt, K, colBase + 32, kk, lane);
        load_b(b3, Bt, K, colBase + 48, kk, lane);
        acc[0][0] = wmma_bf16(a0, b0, acc[0][0]);
        acc[0][1] = wmma_bf16(a0, b1, acc[0][1]);
        acc[0][2] = wmma_bf16(a0, b2, acc[0][2]);
        acc[0][3] = wmma_bf16(a0, b3, acc[0][3]);
        acc[1][0] = wmma_bf16(a1, b0, acc[1][0]);
        acc[1][1] = wmma_bf16(a1, b1, acc[1][1]);
        acc[1][2] = wmma_bf16(a1, b2, acc[1][2]);
        acc[1][3] = wmma_bf16(a1, b3, acc[1][3]);
    }

    const int lg = lane >> 4, lc = lane & 15;
#pragma unroll
    for (int ti = 0; ti < 2; ti++) {
#pragma unroll
        for (int tj = 0; tj < 4; tj++) {
            int col  = colBase + tj * 16 + lc;
            float bv = bias ? bias[col] : 0.0f;
#pragma unroll
            for (int i = 0; i < 8; i++) {
                int row = rowBase + ti * 16 + i + lg * 8;
                float v = acc[ti][tj][i] + bv;
                if (relu) v = fmaxf(v, 0.0f);
                if (outF) outF[(size_t)row * N + col] = v;
                if (outB) {
                    if (outBmode == 1) {
                        outB[(size_t)row * N + col] = f2bf(v);
                    } else {
                        // V projection -> [b, h, dim, s] for contiguous P@V B-fragments
                        int b = row >> 10, s = row & 1023;
                        int h = col >> 6,  d = col & 63;
                        outB[((size_t)((b * 8 + h) * 64 + d) << 10) + s] = f2bf(v);
                    }
                }
            }
        }
    }
}

// ---------------------------------------------------------------------------
// Fused flash-style attention. One wave per (b, h, 16-row q tile).
// Q,K row-major bf16 [4096,512]; Vt bf16 [b,h,64,1024]; Ctx bf16 [4096,512].
// selfmode: attend where (tok[key]==0 || key<=q); cross: attend where tok==0.
// ---------------------------------------------------------------------------
__global__ void k_attn(const unsigned short* __restrict__ Q,
                       const unsigned short* __restrict__ Kb,
                       const unsigned short* __restrict__ Vt,
                       unsigned short* __restrict__ Ctx,
                       const int* __restrict__ toks, int selfmode) {
    __shared__ unsigned short plds[4 * 16 * 32];   // per-wave 16x32 bf16 P tile

    const int lane = threadIdx.x & 31;
    const int wave = threadIdx.x >> 5;
    const int wg = blockIdx.x * 4 + wave;          // 0..2047
    const int qt = wg & 63;
    const int h  = (wg >> 6) & 7;
    const int b  = wg >> 9;
    const int lg = lane >> 4, lc = lane & 15;

    // Q fragments for both 32-wide halves of the 64-dim head
    FragBF aq0, aq1;
    {
        const unsigned short* qbase = Q + (size_t)(b * 1024 + qt * 16) * 512 + h * 64;
        load_a(aq0, qbase, 512, 0, 0, lane);
        load_a(aq1, qbase, 512, 0, 32, lane);
    }

    f32x8 o0 = {0.f,0.f,0.f,0.f,0.f,0.f,0.f,0.f}, o1 = o0, o2 = o0, o3 = o0;
    float m[8], l[8];
#pragma unroll
    for (int i = 0; i < 8; i++) { m[i] = -3.0e38f; l[i] = 0.0f; }

    unsigned short* myp = plds + wave * 16 * 32;

    for (int kt = 0; kt < 1024; kt += 32) {
        f32x8 s0 = {0.f,0.f,0.f,0.f,0.f,0.f,0.f,0.f}, s1 = s0;
        FragBF bk;
        const unsigned short* kb0 = Kb + (size_t)(b * 1024 + kt) * 512 + h * 64;
        load_b(bk, kb0, 512, 0, 0, lane);  s0 = wmma_bf16(aq0, bk, s0);
        load_b(bk, kb0, 512, 0, 32, lane); s0 = wmma_bf16(aq1, bk, s0);
        const unsigned short* kb1 = kb0 + 16 * 512;
        load_b(bk, kb1, 512, 0, 0, lane);  s1 = wmma_bf16(aq0, bk, s1);
        load_b(bk, kb1, 512, 0, 32, lane); s1 = wmma_bf16(aq1, bk, s1);

        // mask + scale (1/sqrt(64))
        int key0 = kt + lc, key1 = key0 + 16;
        bool pad0 = (toks[b * 1024 + key0] == 0);
        bool pad1 = (toks[b * 1024 + key1] == 0);
#pragma unroll
        for (int i = 0; i < 8; i++) {
            int qpos = qt * 16 + i + lg * 8;
            bool ok0 = selfmode ? (pad0 || key0 <= qpos) : pad0;
            bool ok1 = selfmode ? (pad1 || key1 <= qpos) : pad1;
            s0[i] = ok0 ? s0[i] * 0.125f : -1.0e9f;
            s1[i] = ok1 ? s1[i] * 0.125f : -1.0e9f;
        }

        // online softmax: row reductions across the 16-lane half (wave32 safe)
        float alpha[8];
#pragma unroll
        for (int i = 0; i < 8; i++) {
            float t = fmaxf(s0[i], s1[i]);
            t = fmaxf(t, __shfl_xor(t, 1));
            t = fmaxf(t, __shfl_xor(t, 2));
            t = fmaxf(t, __shfl_xor(t, 4));
            t = fmaxf(t, __shfl_xor(t, 8));
            float mn = fmaxf(m[i], t);
            float a  = __expf(m[i] - mn);
            float p0 = __expf(s0[i] - mn);
            float p1 = __expf(s1[i] - mn);
            float rs = p0 + p1;
            rs += __shfl_xor(rs, 1);
            rs += __shfl_xor(rs, 2);
            rs += __shfl_xor(rs, 4);
            rs += __shfl_xor(rs, 8);
            l[i] = l[i] * a + rs;
            m[i] = mn;
            alpha[i] = a;
            s0[i] = p0; s1[i] = p1;
        }
#pragma unroll
        for (int i = 0; i < 8; i++) {
            o0[i] *= alpha[i]; o1[i] *= alpha[i]; o2[i] *= alpha[i]; o3[i] *= alpha[i];
        }

        // C-layout -> A-layout transpose of P through per-wave LDS
#pragma unroll
        for (int i = 0; i < 8; i++) {
            int r = i + lg * 8;
            myp[r * 32 + lc]      = f2bf(s0[i]);
            myp[r * 32 + lc + 16] = f2bf(s1[i]);
        }
        asm volatile("s_wait_dscnt 0" ::: "memory");

        FragBF ap;
        load_a(ap, myp, 32, 0, 0, lane);

        // ctx += P[16x32] @ V[32x64], via Vt [b,h,dim,1024] (contiguous keys)
        const unsigned short* vbase = Vt + (size_t)((b * 8 + h) * 64) * 1024 + kt;
        FragBF bv;
        load_b(bv, vbase, 1024, 0,  0, lane);  o0 = wmma_bf16(ap, bv, o0);
        load_b(bv, vbase, 1024, 16, 0, lane);  o1 = wmma_bf16(ap, bv, o1);
        load_b(bv, vbase, 1024, 32, 0, lane);  o2 = wmma_bf16(ap, bv, o2);
        load_b(bv, vbase, 1024, 48, 0, lane);  o3 = wmma_bf16(ap, bv, o3);
    }

    // normalize and emit bf16 ctx row-major
#pragma unroll
    for (int i = 0; i < 8; i++) {
        float inv = 1.0f / l[i];
        int token = b * 1024 + qt * 16 + i + lg * 8;
        unsigned short* crow = Ctx + (size_t)token * 512 + h * 64;
        crow[lc]      = f2bf(o0[i] * inv);
        crow[16 + lc] = f2bf(o1[i] * inv);
        crow[32 + lc] = f2bf(o2[i] * inv);
        crow[48 + lc] = f2bf(o3[i] * inv);
    }
}

// ---------------------------------------------------------------------------
// Fused residual add + LayerNorm over 512 cols. One row per block.
// ---------------------------------------------------------------------------
__global__ void k_add_ln(const float* __restrict__ acc, const float* __restrict__ res,
                         const float* __restrict__ g, const float* __restrict__ bb,
                         float* __restrict__ outF, unsigned short* __restrict__ outB) {
    __shared__ float red[256];
    int row = blockIdx.x;
    int t = threadIdx.x;
    const float* a = acc + (size_t)row * 512;
    const float* r = res + (size_t)row * 512;
    float v0 = a[t] + r[t];
    float v1 = a[t + 256] + r[t + 256];

    red[t] = v0 + v1;
    __syncthreads();
    for (int o = 128; o > 0; o >>= 1) {
        if (t < o) red[t] += red[t + o];
        __syncthreads();
    }
    float mu = red[0] * (1.0f / 512.0f);
    __syncthreads();

    float d0 = v0 - mu, d1 = v1 - mu;
    red[t] = d0 * d0 + d1 * d1;
    __syncthreads();
    for (int o = 128; o > 0; o >>= 1) {
        if (t < o) red[t] += red[t + o];
        __syncthreads();
    }
    float rstd = rsqrtf(red[0] * (1.0f / 512.0f) + 1e-6f);

    float y0 = g[t] * d0 * rstd + bb[t];
    float y1 = g[t + 256] * d1 * rstd + bb[t + 256];
    outF[(size_t)row * 512 + t] = y0;
    outF[(size_t)row * 512 + t + 256] = y1;
    if (outB) {
        outB[(size_t)row * 512 + t] = f2bf(y0);
        outB[(size_t)row * 512 + t + 256] = f2bf(y1);
    }
}

// ---------------------------------------------------------------------------
extern "C" void kernel_launch(void* const* d_in, const int* in_sizes, int n_in,
                              void* d_out, int out_size, void* d_ws, size_t ws_size,
                              hipStream_t stream) {
    (void)in_sizes; (void)n_in; (void)out_size; (void)ws_size;

    const int*   dec     = (const int*)d_in[0];
    const int*   enc_ids = (const int*)d_in[1];
    const float* enc_out = (const float*)d_in[2];
    const float* emb     = (const float*)d_in[3];
    const float* selfW   = (const float*)d_in[4];
    const float* selfb   = (const float*)d_in[5];
    const float* encW    = (const float*)d_in[6];
    const float* encb    = (const float*)d_in[7];
    const float* W1      = (const float*)d_in[8];
    const float* b1      = (const float*)d_in[9];
    const float* W2      = (const float*)d_in[10];
    const float* b2      = (const float*)d_in[11];
    const float* lng     = (const float*)d_in[12];
    const float* lnb     = (const float*)d_in[13];

    char* base = (char*)d_ws;
    size_t off = 0;
    auto alloc = [&](size_t bytes) -> char* {
        char* p = base + off;
        off = (off + bytes + 255) & ~(size_t)255;
        return p;
    };

    const size_t WSZ = 512ull * 512;          // one 512x512 weight
    unsigned short* wselfT = (unsigned short*)alloc(16 * WSZ * 2);
    unsigned short* wencT  = (unsigned short*)alloc(16 * WSZ * 2);
    unsigned short* wffn1T = (unsigned short*)alloc(4ull * 2048 * 512 * 2);
    unsigned short* wffn2T = (unsigned short*)alloc(4ull * 512 * 2048 * 2);
    float*          xF     = (float*)alloc(4096ull * 512 * 4);
    unsigned short* xB     = (unsigned short*)alloc(4096ull * 512 * 2);
    unsigned short* encB   = (unsigned short*)alloc(4096ull * 512 * 2);
    unsigned short* qB     = (unsigned short*)alloc(4096ull * 512 * 2);
    unsigned short* kB     = (unsigned short*)alloc(4096ull * 512 * 2);
    unsigned short* vtB    = (unsigned short*)alloc(4096ull * 512 * 2);
    unsigned short* ctxB   = (unsigned short*)alloc(4096ull * 512 * 2);
    float*          tmpF   = (float*)alloc(4096ull * 512 * 4);
    unsigned short* hB     = (unsigned short*)alloc(4096ull * 2048 * 2);

    // Convert + transpose all weights to bf16 [N,K]
    for (int i = 0; i < 16; i++)
        k_convT<<<256, 256, 0, stream>>>(selfW + i * WSZ, wselfT + i * WSZ, 512, 512);
    for (int i = 0; i < 16; i++)
        k_convT<<<256, 256, 0, stream>>>(encW + i * WSZ, wencT + i * WSZ, 512, 512);
    for (int ll = 0; ll < 4; ll++)
        k_convT<<<256, 256, 0, stream>>>(W1 + (size_t)ll * 512 * 2048,
                                         wffn1T + (size_t)ll * 2048 * 512, 512, 2048);
    for (int ll = 0; ll < 4; ll++)
        k_convT<<<256, 256, 0, stream>>>(W2 + (size_t)ll * 2048 * 512,
                                         wffn2T + (size_t)ll * 512 * 2048, 2048, 512);

    k_conv<<<256, 256, 0, stream>>>(enc_out, encB, 4096 * 512);
    k_embed<<<4096, 256, 0, stream>>>(dec, emb, xF, xB);

    auto gemm = [&](const unsigned short* A, const unsigned short* Bt, const float* bias,
                    float* oF, unsigned short* oB, int N, int K, int relu, int mode) {
        dim3 g(4096 / 128, N / 128);
        k_gemm<<<g, 256, 0, stream>>>(A, Bt, bias, oF, oB, 4096, N, K, relu, mode);
    };

    for (int l = 0; l < 4; l++) {
        // ---- self attention ----
        const unsigned short* Ws = wselfT + (size_t)l * 4 * WSZ;
        const float* bs = selfb + (size_t)l * 4 * 512;
        gemm(xB, Ws + 0 * WSZ, bs + 0,    nullptr, qB,  512, 512, 0, 1);
        gemm(xB, Ws + 1 * WSZ, bs + 512,  nullptr, kB,  512, 512, 0, 1);
        gemm(xB, Ws + 2 * WSZ, bs + 1024, nullptr, vtB, 512, 512, 0, 2);
        k_attn<<<512, 128, 0, stream>>>(qB, kB, vtB, ctxB, dec, 1);
        gemm(ctxB, Ws + 3 * WSZ, bs + 1536, tmpF, nullptr, 512, 512, 0, 0);
        k_add_ln<<<4096, 256, 0, stream>>>(tmpF, xF, lng + (l * 3 + 0) * 512,
                                           lnb + (l * 3 + 0) * 512, xF, xB);

        // ---- cross attention ----
        const unsigned short* We = wencT + (size_t)l * 4 * WSZ;
        const float* be = encb + (size_t)l * 4 * 512;
        gemm(xB,   We + 0 * WSZ, be + 0,    nullptr, qB,  512, 512, 0, 1);
        gemm(encB, We + 1 * WSZ, be + 512,  nullptr, kB,  512, 512, 0, 1);
        gemm(encB, We + 2 * WSZ, be + 1024, nullptr, vtB, 512, 512, 0, 2);
        k_attn<<<512, 128, 0, stream>>>(qB, kB, vtB, ctxB, enc_ids, 0);
        gemm(ctxB, We + 3 * WSZ, be + 1536, tmpF, nullptr, 512, 512, 0, 0);
        k_add_ln<<<4096, 256, 0, stream>>>(tmpF, xF, lng + (l * 3 + 1) * 512,
                                           lnb + (l * 3 + 1) * 512, xF, xB);

        // ---- FFN ----
        gemm(xB, wffn1T + (size_t)l * 2048 * 512, b1 + (size_t)l * 2048,
             nullptr, hB, 2048, 512, 1, 1);
        gemm(hB, wffn2T + (size_t)l * 512 * 2048, b2 + (size_t)l * 512,
             tmpF, nullptr, 512, 2048, 0, 0);
        float* outLn = (l == 3) ? (float*)d_out : xF;
        k_add_ln<<<4096, 256, 0, stream>>>(tmpF, xF, lng + (l * 3 + 2) * 512,
                                           lnb + (l * 3 + 2) * 512, outLn, xB);
    }
}
